// GINPolicyNetwork_8040178778546
// MI455X (gfx1250) — compile-verified, matrix operand order checked by reading, not run on hardware
//
#include <hip/hip_runtime.h>

// ---------------------------------------------------------------------------
// GIN policy network forward, CDNA5 (gfx1250, wave32, WMMA f16 -> f32 acc).
// B=64, N=512 nodes, H=256, L=2 transformer layers, O=5.  T = 32768 tokens.
// Dense layers: register-blocked WMMA GEMM, 32x32 output tile per wave
// (2x2 accumulators -> 2 b128 loads per v_wmma instead of 4).
// Graph aggregation X + M@X: WMMA with A synthesized from a 512-bit row
// bitmask and B read from a pre-transposed activation buffer (contiguous
// 16B fragment loads instead of strided u16 gathers).
// ---------------------------------------------------------------------------

typedef __attribute__((ext_vector_type(16))) _Float16     v16h;
typedef __attribute__((ext_vector_type(8)))  float        v8f;
typedef __attribute__((ext_vector_type(4)))  unsigned int u32x4;

union FragH { v16h v; u32x4 u[2]; };

#define TOKENS 32768
#define BATCH  64
#define NODES  512
#define HID    256

#define WMMA_F16(A, B, C) \
    __builtin_amdgcn_wmma_f32_16x16x32_f16(false, (A), false, (B), (short)0, (C), false, false)

// ---------------------------------------------------------------------------
// Lidar MLP (20->256->20) + assemble node features as f16, K padded 38->64.
// One block (256 thr) per token.
// ---------------------------------------------------------------------------
__global__ void k_node(const float* __restrict__ data,
                       const float* __restrict__ lw1, const float* __restrict__ lb1,
                       const float* __restrict__ lw2, const float* __restrict__ lb2,
                       _Float16* __restrict__ node)
{
    __shared__ float lid[20];
    __shared__ float hid[256];
    __shared__ float lf2[20];
    size_t t = blockIdx.x;
    int tid = threadIdx.x;
    if (tid < 20) lid[tid] = data[t * 38 + 18 + tid];
    __syncthreads();
    float s = lb1[tid];
    for (int k = 0; k < 20; ++k) s += lid[k] * lw1[tid * 20 + k];
    hid[tid] = fmaxf(s, 0.0f);
    __syncthreads();
    if (tid < 20) {
        float s2 = lb2[tid];
        for (int k = 0; k < 256; ++k) s2 += hid[k] * lw2[tid * 256 + k];
        lf2[tid] = fmaxf(s2, 0.0f);
    }
    __syncthreads();
    if (tid < 64) {
        float v = (tid < 18) ? data[t * 38 + tid]
                             : ((tid < 38) ? lf2[tid - 18] : 0.0f);
        node[t * 64 + tid] = (_Float16)v;
    }
}

// ---------------------------------------------------------------------------
// Adjacency bitmask: bit j of row i (per batch) = (i!=j) && dist^2 <= 100.
// (|ang|<=pi is always true after wrapping -> mask is distance-only, and
// dist is symmetric so M^T = M.)
// 8 waves/block, one wave per row; lane l covers j = l*16 .. l*16+15.
// Stored as 32 u16 words per row (u32 word w covers k-range [32w, 32w+31]).
// ---------------------------------------------------------------------------
__global__ void k_mask(const float* __restrict__ data,
                       unsigned short* __restrict__ mask)
{
    int gid = blockIdx.x;
    int b = gid >> 6;
    int i = ((gid & 63) << 3) + (threadIdx.x >> 5);
    int lane = threadIdx.x & 31;
    size_t base = (size_t)b << 9;
    float px = data[(base + i) * 38 + 0];
    float py = data[(base + i) * 38 + 1];
    unsigned int bits = 0;
    for (int t = 0; t < 16; ++t) {
        int j = lane * 16 + t;
        float dx = data[(base + j) * 38 + 0] - px;
        float dy = data[(base + j) * 38 + 1] - py;
        float d2 = dx * dx + dy * dy;
        if ((j != i) && (d2 <= 100.0f)) bits |= (1u << t);
    }
    mask[(base + i) * 32 + lane] = (unsigned short)bits;
}

// ---------------------------------------------------------------------------
// Weight convert f32 -> f16, row-major [rows x kout], zero-pad kin -> kout.
// ---------------------------------------------------------------------------
__global__ void k_cvt(const float* __restrict__ in, _Float16* __restrict__ out,
                      int rows, int kin, int kout)
{
    int idx = blockIdx.x * 256 + threadIdx.x;
    if (idx >= rows * kout) return;
    int r = idx / kout, c = idx % kout;
    out[idx] = (c < kin) ? (_Float16)in[r * kin + c] : (_Float16)0.0f;
}

// ---------------------------------------------------------------------------
// Activation transpose for aggregation: X [T x C] -> XT [B][C][512].
// ---------------------------------------------------------------------------
__global__ void k_tr(const _Float16* __restrict__ in, _Float16* __restrict__ out,
                     int C)
{
    size_t idx = (size_t)blockIdx.x * 256 + threadIdx.x;
    size_t t = idx / C;
    int    c = (int)(idx % C);
    int    b = (int)(t >> 9);
    int    k = (int)(t & 511);
    out[((size_t)b * C + c) * 512 + k] = in[idx];
}

// ---------------------------------------------------------------------------
// WMMA GEMM:  out[M x N] = act( A[M x K] @ W[N x K]^T + bias (+ res) )
// One wave per 32x32 output tile (2x2 WMMA accumulators).
// A-fragment: lane m=lane&15 holds row m, K runs {koff..+7, koff+16..+23},
// koff = 8*(lane>>4) (ISA 7.12.2).  W stored [N x K] row-major => B-fragment
// load pattern is identical.  C/D: col = lane&15, row = vgpr + 8*(lane>>4).
// ---------------------------------------------------------------------------
__global__ void k_gemm(const _Float16* __restrict__ A,
                       const _Float16* __restrict__ W,
                       const float*    __restrict__ bias,
                       const _Float16* __restrict__ res,
                       _Float16*       __restrict__ out,
                       int M, int K, int N, int relu)
{
    int wid    = blockIdx.x * 8 + (threadIdx.x >> 5);
    int tilesN = N >> 5;                       // 32-wide groups
    int tm = wid / tilesN;
    int tn = wid % tilesN;
    if (tm * 32 >= M) return;                  // wave-uniform
    int lane = threadIdx.x & 31;
    int hf   = lane >> 4;
    int idx  = lane & 15;

    const _Float16* ar0 = A + (size_t)(tm * 32 + idx) * K + hf * 8;
    const _Float16* ar1 = ar0 + (size_t)16 * K;
    const _Float16* br0 = W + (size_t)(tn * 32 + idx) * K + hf * 8;
    const _Float16* br1 = br0 + (size_t)16 * K;

    FragH fa0, fa1, fb0, fb1;
    v8f a00 = {}, a01 = {}, a10 = {}, a11 = {};
    for (int k0 = 0; k0 < K; k0 += 32) {
        fa0.u[0] = *(const u32x4*)(ar0 + k0);
        fa0.u[1] = *(const u32x4*)(ar0 + k0 + 16);
        fa1.u[0] = *(const u32x4*)(ar1 + k0);
        fa1.u[1] = *(const u32x4*)(ar1 + k0 + 16);
        fb0.u[0] = *(const u32x4*)(br0 + k0);
        fb0.u[1] = *(const u32x4*)(br0 + k0 + 16);
        fb1.u[0] = *(const u32x4*)(br1 + k0);
        fb1.u[1] = *(const u32x4*)(br1 + k0 + 16);
        a00 = WMMA_F16(fa0.v, fb0.v, a00);
        a01 = WMMA_F16(fa0.v, fb1.v, a01);
        a10 = WMMA_F16(fa1.v, fb0.v, a10);
        a11 = WMMA_F16(fa1.v, fb1.v, a11);
    }

    int col0 = tn * 32 + idx;
    int col1 = col0 + 16;
    float bv0 = bias ? bias[col0] : 0.0f;
    float bv1 = bias ? bias[col1] : 0.0f;
    for (int r = 0; r < 8; ++r) {
        int row0 = tm * 32 + r + hf * 8;
        int row1 = row0 + 16;
        float v00 = a00[r] + bv0;
        float v01 = a01[r] + bv1;
        float v10 = a10[r] + bv0;
        float v11 = a11[r] + bv1;
        if (res) {
            v00 += (float)res[(size_t)row0 * N + col0];
            v01 += (float)res[(size_t)row0 * N + col1];
            v10 += (float)res[(size_t)row1 * N + col0];
            v11 += (float)res[(size_t)row1 * N + col1];
        }
        if (relu) {
            v00 = fmaxf(v00, 0.0f); v01 = fmaxf(v01, 0.0f);
            v10 = fmaxf(v10, 0.0f); v11 = fmaxf(v11, 0.0f);
        }
        out[(size_t)row0 * N + col0] = (_Float16)v00;
        out[(size_t)row0 * N + col1] = (_Float16)v01;
        out[(size_t)row1 * N + col0] = (_Float16)v10;
        out[(size_t)row1 * N + col1] = (_Float16)v11;
    }
}

// ---------------------------------------------------------------------------
// Graph aggregation via WMMA: out[b] = X[b] + M_b @ X[b]  (M_b symmetric 0/1)
// One wave per (batch, 16 j-rows, 32 c-cols) -> mask A-fragment expanded once
// per k-tile, reused for 2 WMMAs.  B fragments come from transposed XT with
// contiguous 16B loads.  K = 512 -> 16 k-tiles.
// ---------------------------------------------------------------------------
__global__ void k_agg(const _Float16* __restrict__ X,      // [b,k,C] row-major
                      const _Float16* __restrict__ XT,     // [b,C,512]
                      const unsigned short* __restrict__ mask,
                      _Float16* __restrict__ out, int C)
{
    int wid    = blockIdx.x * 8 + (threadIdx.x >> 5);
    int cTiles = C >> 5;                       // 32-wide c groups
    int tpb    = 32 * cTiles;                  // (512/16) j-tiles * c-groups
    int b  = wid / tpb;
    int r  = wid % tpb;
    int tj = r / cTiles;
    int tc = r % cTiles;
    int lane = threadIdx.x & 31;
    int hf   = lane >> 4;
    int idx  = lane & 15;
    int j    = tj * 16 + idx;
    int c0   = tc * 32 + idx;
    int c1   = c0 + 16;

    const unsigned short* mrow = mask + (((size_t)b << 9) + j) * 32;
    const _Float16* bt0 = XT + ((size_t)b * C + c0) * 512 + hf * 8;
    const _Float16* bt1 = XT + ((size_t)b * C + c1) * 512 + hf * 8;
    const _Float16* Xb  = X + ((size_t)b << 9) * C;

    FragH fa, fb0, fb1;
    v8f acc0 = {}, acc1 = {};
    for (int kt = 0; kt < 16; ++kt) {
        int k0 = kt * 32;
        unsigned int w = (unsigned int)mrow[kt * 2]
                       | ((unsigned int)mrow[kt * 2 + 1] << 16);
        for (int i = 0; i < 8; ++i) {
            fa.v[i]     = ((w >> (hf * 8 + i))      & 1) ? (_Float16)1.0f : (_Float16)0.0f;
            fa.v[i + 8] = ((w >> (hf * 8 + 16 + i)) & 1) ? (_Float16)1.0f : (_Float16)0.0f;
        }
        fb0.u[0] = *(const u32x4*)(bt0 + k0);
        fb0.u[1] = *(const u32x4*)(bt0 + k0 + 16);
        fb1.u[0] = *(const u32x4*)(bt1 + k0);
        fb1.u[1] = *(const u32x4*)(bt1 + k0 + 16);
        acc0 = WMMA_F16(fa.v, fb0.v, acc0);
        acc1 = WMMA_F16(fa.v, fb1.v, acc1);
    }
    for (int rr = 0; rr < 8; ++rr) {
        int row = tj * 16 + rr + hf * 8;
        size_t o = ((size_t)(b << 9) + row) * (size_t)C;
        out[o + c0] = (_Float16)((float)Xb[(size_t)row * C + c0] + acc0[rr]);
        out[o + c1] = (_Float16)((float)Xb[(size_t)row * C + c1] + acc1[rr]);
    }
}

// ---------------------------------------------------------------------------
// LayerNorm over C=256, one block per token (in-place safe).
// ---------------------------------------------------------------------------
__global__ void k_ln(const _Float16* __restrict__ x,
                     const float* __restrict__ g, const float* __restrict__ bta,
                     _Float16* __restrict__ out)
{
    __shared__ float red[256];
    size_t t = blockIdx.x;
    int tid = threadIdx.x;
    float v = (float)x[t * 256 + tid];
    red[tid] = v; __syncthreads();
    for (int s = 128; s > 0; s >>= 1) { if (tid < s) red[tid] += red[tid + s]; __syncthreads(); }
    float m = red[0] * (1.0f / 256.0f);
    __syncthreads();
    float d = v - m;
    red[tid] = d * d; __syncthreads();
    for (int s = 128; s > 0; s >>= 1) { if (tid < s) red[tid] += red[tid + s]; __syncthreads(); }
    float var = red[0] * (1.0f / 256.0f);
    float y = d * rsqrtf(var + 1e-5f) * g[tid] + bta[tid];
    out[t * 256 + tid] = (_Float16)y;
}

// ---------------------------------------------------------------------------
// Final head 256 -> 5, f32 output.
// ---------------------------------------------------------------------------
__global__ void k_fc(const _Float16* __restrict__ h, const float* __restrict__ fcw,
                     const float* __restrict__ fcb, float* __restrict__ out, int total)
{
    int idx = blockIdx.x * 256 + threadIdx.x;
    if (idx >= total) return;
    int t = idx / 5, o = idx % 5;
    float s = fcb[o];
    const _Float16* hr = h + (size_t)t * 256;
    const float*    wr = fcw + o * 256;
    for (int k = 0; k < 256; ++k) s += (float)hr[k] * wr[k];
    out[idx] = s;
}

// ---------------------------------------------------------------------------
extern "C" void kernel_launch(void* const* d_in, const int* in_sizes, int n_in,
                              void* d_out, int out_size, void* d_ws, size_t ws_size,
                              hipStream_t stream)
{
    const int H = HID, T = TOKENS;
    const float* data    = (const float*)d_in[0];
    const float* lw1     = (const float*)d_in[1];
    const float* lb1     = (const float*)d_in[2];
    const float* lw2     = (const float*)d_in[3];
    const float* lb2     = (const float*)d_in[4];
    const float* g1w1    = (const float*)d_in[5];
    const float* g1b1    = (const float*)d_in[6];
    const float* g1w2    = (const float*)d_in[7];
    const float* g1b2    = (const float*)d_in[8];
    const float* g2w1    = (const float*)d_in[9];
    const float* g2b1    = (const float*)d_in[10];
    const float* g2w2    = (const float*)d_in[11];
    const float* g2b2    = (const float*)d_in[12];
    const float* t_in_w  = (const float*)d_in[13];
    const float* t_in_b  = (const float*)d_in[14];
    const float* t_out_w = (const float*)d_in[15];
    const float* t_out_b = (const float*)d_in[16];
    const float* t_ln1_g = (const float*)d_in[17];
    const float* t_ln1_b = (const float*)d_in[18];
    const float* t_ff1_w = (const float*)d_in[19];
    const float* t_ff1_b = (const float*)d_in[20];
    const float* t_ff2_w = (const float*)d_in[21];
    const float* t_ff2_b = (const float*)d_in[22];
    const float* t_ln2_g = (const float*)d_in[23];
    const float* t_ln2_b = (const float*)d_in[24];
    const float* fc_w    = (const float*)d_in[25];
    const float* fc_b    = (const float*)d_in[26];

    // Workspace carve-up (~78 MB total).
    char* ws = (char*)d_ws;
    size_t off = 0;
    auto alloc = [&](size_t bytes) -> char* {
        char* p = ws + off;
        off = (off + bytes + 255) & ~(size_t)255;
        return p;
    };
    _Float16* node    = (_Float16*)alloc((size_t)T * 64 * 2);
    _Float16* nodeAgg = (_Float16*)alloc((size_t)T * 64 * 2);
    unsigned short* mbits = (unsigned short*)alloc((size_t)T * 32 * 2);
    _Float16* X0 = (_Float16*)alloc((size_t)T * H * 2);
    _Float16* X1 = (_Float16*)alloc((size_t)T * H * 2);
    _Float16* X2 = (_Float16*)alloc((size_t)T * H * 2);
    _Float16* XT = (_Float16*)alloc((size_t)T * H * 2);   // transpose scratch
    _Float16* wg1w1 = (_Float16*)alloc((size_t)256 * 64 * 2);
    _Float16* wg1w2 = (_Float16*)alloc((size_t)H * H * 2);
    _Float16* wg2w1 = (_Float16*)alloc((size_t)H * H * 2);
    _Float16* wg2w2 = (_Float16*)alloc((size_t)H * H * 2);
    _Float16 *wWv[2], *wOut[2], *wF1[2], *wF2[2];
    for (int l = 0; l < 2; ++l) {
        wWv[l]  = (_Float16*)alloc((size_t)H * H * 2);
        wOut[l] = (_Float16*)alloc((size_t)H * H * 2);
        wF1[l]  = (_Float16*)alloc((size_t)H * H * 2);
        wF2[l]  = (_Float16*)alloc((size_t)H * H * 2);
    }

    auto cvt = [&](const float* in, _Float16* outp, int rows, int kin, int kout) {
        int n = rows * kout;
        k_cvt<<<dim3((n + 255) / 256), dim3(256), 0, stream>>>(in, outp, rows, kin, kout);
    };
    auto gemm = [&](const _Float16* A, const _Float16* W, const float* bias,
                    const _Float16* res, _Float16* outp, int K, int Nout, int relu) {
        int waves = (T / 32) * (Nout / 32);
        k_gemm<<<dim3(waves / 8), dim3(256), 0, stream>>>(A, W, bias, res, outp,
                                                          T, K, Nout, relu);
    };
    auto agg = [&](const _Float16* X, _Float16* outp, int C) {
        int ntr = (T * C) / 256;
        k_tr<<<dim3(ntr), dim3(256), 0, stream>>>(X, XT, C);
        int waves = BATCH * 32 * (C / 32);
        k_agg<<<dim3(waves / 8), dim3(256), 0, stream>>>(X, XT, mbits, outp, C);
    };

    // Stage 0: node features + adjacency bits + weight conversion.
    k_node<<<dim3(T), dim3(256), 0, stream>>>(data, lw1, lb1, lw2, lb2, node);
    k_mask<<<dim3(BATCH * 64), dim3(256), 0, stream>>>(data, mbits);
    cvt(g1w1, wg1w1, 256, 38, 64);
    cvt(g1w2, wg1w2, H, H, H);
    cvt(g2w1, wg2w1, H, H, H);
    cvt(g2w2, wg2w2, H, H, H);
    for (int l = 0; l < 2; ++l) {
        cvt(t_in_w + (size_t)l * 3 * H * H + 2 * H * H, wWv[l], H, H, H); // Wv rows [2H,3H)
        cvt(t_out_w + (size_t)l * H * H, wOut[l], H, H, H);
        cvt(t_ff1_w + (size_t)l * H * H, wF1[l], H, H, H);
        cvt(t_ff2_w + (size_t)l * H * H, wF2[l], H, H, H);
    }

    // GIN block 1:  h = relu(relu((node + M@node) @ g1w1^T + b) @ g1w2^T + b)
    agg(node, nodeAgg, 64);
    gemm(nodeAgg, wg1w1, g1b1, nullptr, X1, 64, H, 1);
    gemm(X1, wg1w2, g1b2, nullptr, X0, H, H, 1);

    // GIN block 2:  h = relu(relu((h + M@h) @ g2w1^T + b) @ g2w2^T + b)
    agg(X0, X1, H);
    gemm(X1, wg2w1, g2b1, nullptr, X2, H, H, 1);
    gemm(X2, wg2w2, g2b2, nullptr, X0, H, H, 1);

    // "Transformer" layers: two linear ops + residual + LN, then FFN + LN.
    for (int l = 0; l < 2; ++l) {
        gemm(X0, wWv[l],  t_in_b + (size_t)l * 3 * H + 2 * H, nullptr, X1, H, H, 0);
        gemm(X1, wOut[l], t_out_b + (size_t)l * H,            X0,      X2, H, H, 0);
        k_ln<<<dim3(T), dim3(256), 0, stream>>>(X2, t_ln1_g + (size_t)l * H,
                                                t_ln1_b + (size_t)l * H, X1);
        gemm(X1, wF1[l], t_ff1_b + (size_t)l * H, nullptr, X2, H, H, 1);
        gemm(X2, wF2[l], t_ff2_b + (size_t)l * H, X1,      X0, H, H, 0);
        k_ln<<<dim3(T), dim3(256), 0, stream>>>(X0, t_ln2_g + (size_t)l * H,
                                                t_ln2_b + (size_t)l * H, X0);
    }

    // Head.
    k_fc<<<dim3((T * 5 + 255) / 256), dim3(256), 0, stream>>>(X0, fc_w, fc_b,
                                                              (float*)d_out, T * 5);
}
